// DynamicConv_16819091931665
// MI455X (gfx1250) — matrix operand order repaired and checked by reading, not compiled, not run on hardware
//
#include <hip/hip_runtime.h>
#include <hip/hip_bf16.h>

typedef __bf16 v16bf __attribute__((ext_vector_type(16)));
typedef float  v8f   __attribute__((ext_vector_type(8)));
typedef int    v4i   __attribute__((ext_vector_type(4)));

struct __attribute__((aligned(16))) U4 { unsigned int x, y, z, w; };

#define BN 32
#define CI 128
#define CO 128
#define KB 4
#define HH 64
#define WW 64
#define HW 4096
#define TAPS 25

// ---- CDNA5 async global->LDS path (guarded: falls back to sync copies) ----
#if defined(__has_builtin)
#if __has_builtin(__builtin_amdgcn_global_load_async_to_lds_b128) && \
    __has_builtin(__builtin_amdgcn_s_wait_asynccnt)
#define HAVE_ASYNC_LDS 1
#endif
#endif

__device__ __forceinline__ void async_copy16(const void* g, void* l) {
#ifdef HAVE_ASYNC_LDS
    __builtin_amdgcn_global_load_async_to_lds_b128(
        (__attribute__((address_space(1))) v4i*)(void*)g,
        (__attribute__((address_space(3))) v4i*)l,
        0, 0);
#else
    *(U4*)l = *(const U4*)g;
#endif
}

__device__ __forceinline__ void async_join() {
#ifdef HAVE_ASYNC_LDS
    __builtin_amdgcn_s_wait_asynccnt(0);
#endif
}

// ---------------------------------------------------------------------------
// 1) global average pool: pooled[b*CI+c] = mean over HW
// ---------------------------------------------------------------------------
__global__ void pool_kernel(const float* __restrict__ x, float* __restrict__ pooled) {
    __shared__ float red[256];
    const int bc = blockIdx.x;                 // b*CI + c
    const float* p = x + (size_t)bc * HW;
    float s = 0.f;
    for (int i = threadIdx.x; i < HW; i += 256) s += p[i];
    red[threadIdx.x] = s;
    __syncthreads();
    for (int off = 128; off > 0; off >>= 1) {
        if (threadIdx.x < off) red[threadIdx.x] += red[threadIdx.x + off];
        __syncthreads();
    }
    if (threadIdx.x == 0) pooled[bc] = red[0] * (1.0f / (float)HW);
}

// ---------------------------------------------------------------------------
// 2) dense -> relu -> dense -> softmax  + effective bias  (tiny; 1 block / b)
// ---------------------------------------------------------------------------
__global__ void attn_kernel(const float* __restrict__ pooled,
                            const float* __restrict__ a1w, const float* __restrict__ a1b,
                            const float* __restrict__ a2w, const float* __restrict__ a2b,
                            const float* __restrict__ bias,
                            float* __restrict__ attn, float* __restrict__ bias_eff) {
    __shared__ float hbuf[CI];
    __shared__ float lg[KB];
    const int b = blockIdx.x, t = threadIdx.x;     // t in [0,128)
    float s = a1b[t];
    for (int i = 0; i < CI; ++i) s += pooled[b * CI + i] * a1w[t * CI + i];
    hbuf[t] = fmaxf(s, 0.f);
    __syncthreads();
    if (t < KB) {
        float l = a2b[t];
        for (int i = 0; i < CI; ++i) l += hbuf[i] * a2w[t * CI + i];
        lg[t] = l;
    }
    __syncthreads();
    float m = fmaxf(fmaxf(lg[0], lg[1]), fmaxf(lg[2], lg[3]));
    float e[KB];
    float sum = 0.f;
    for (int k = 0; k < KB; ++k) { e[k] = expf(lg[k] - m); sum += e[k]; }
    const float inv = 1.0f / sum;
    if (t < KB) attn[b * KB + t] = e[t] * inv;
    float be = 0.f;
    for (int k = 0; k < KB; ++k) be += e[k] * inv * bias[k * CO + t];
    bias_eff[b * CO + t] = be;
}

// ---------------------------------------------------------------------------
// 3) x: NCHW f32 -> NHWC bf16 (LDS-tiled transpose)
//    xh[(b*HW + p)*CI + c]
// ---------------------------------------------------------------------------
__global__ void convert_x(const float* __restrict__ x, __hip_bfloat16* __restrict__ xh) {
    __shared__ float lds[64 * 33];
    const int blk = blockIdx.x;          // b*(64*4) + pt*4 + cb
    const int cb = blk & 3;
    const int pt = (blk >> 2) & 63;
    const int b  = blk >> 8;
    const int c0 = cb * 32, p0 = pt * 64;
    const int t = threadIdx.x;
    for (int j = 0; j < 8; ++j) {
        int e = t + j * 256;             // 2048 elements: 32 c x 64 p
        int cl = e >> 6, pl = e & 63;
        lds[pl * 33 + cl] = x[((size_t)(b * CI + c0 + cl)) * HW + p0 + pl];
    }
    __syncthreads();
    for (int j = 0; j < 8; ++j) {
        int e = t + j * 256;
        int pl = e >> 5, cl = e & 31;
        xh[((size_t)(b * HW + p0 + pl)) * CI + c0 + cl] = __float2bfloat16(lds[pl * 33 + cl]);
    }
}

// ---------------------------------------------------------------------------
// 4) weight (K,O,I,5,5) f32 -> wt[k][tap][o][i] f32 (contiguous i)
// ---------------------------------------------------------------------------
__global__ void transpose_w(const float* __restrict__ weight, float* __restrict__ wt) {
    size_t idx = (size_t)blockIdx.x * 256 + threadIdx.x;   // total 4*25*128*128
    if (idx >= (size_t)KB * TAPS * CO * CI) return;
    int i = idx & 127;
    size_t r1 = idx >> 7;
    int o = r1 & 127;
    size_t r2 = r1 >> 7;
    int tap = (int)(r2 % TAPS);
    int k = (int)(r2 / TAPS);
    wt[idx] = weight[(((size_t)(k * CO + o) * CI + i) * TAPS) + tap];
}

// ---------------------------------------------------------------------------
// 5) per-sample effective weights in bf16:
//    weff[((tap*CO + o)*BN + b)*CI + i] = sum_k attn[b,k]*wt[k][tap][o][i]
// ---------------------------------------------------------------------------
__global__ void combine_w(const float* __restrict__ wt, const float* __restrict__ attn,
                          __hip_bfloat16* __restrict__ weff) {
    const int to = blockIdx.x;           // tap*CO + o, [0,3200)
    const int i = threadIdx.x;           // [0,128)
    const int tap = to >> 7, o = to & 127;
    const size_t kstride = (size_t)TAPS * CO * CI;
    const size_t base = ((size_t)tap * CO + o) * CI + i;
    float w0 = wt[base];
    float w1 = wt[base + kstride];
    float w2 = wt[base + 2 * kstride];
    float w3 = wt[base + 3 * kstride];
    for (int b = 0; b < BN; ++b) {
        float a0 = attn[b * 4 + 0], a1 = attn[b * 4 + 1];
        float a2 = attn[b * 4 + 2], a3 = attn[b * 4 + 3];
        float v = a0 * w0 + a1 * w1 + a2 * w2 + a3 * w3;
        weff[(((size_t)to * BN) + b) * CI + i] = __float2bfloat16(v);
    }
}

// ---------------------------------------------------------------------------
// 6) implicit-GEMM conv with per-sample weights, bf16 WMMA, f32 accumulate.
//    Block: 1 sample x 2 output rows (128 pixels) x 128 output channels.
//    8 waves: wave grid 4 (M) x 2 (N); wave = 2 M-tiles x 4 N-tiles.
//    Staging uses GLOBAL_LOAD_ASYNC_TO_LDS_B128 (ASYNCcnt) when available.
// ---------------------------------------------------------------------------
__global__ void conv_wmma(const __hip_bfloat16* __restrict__ xh,
                          const __hip_bfloat16* __restrict__ weff,
                          const float* __restrict__ bias_eff,
                          float* __restrict__ out) {
    extern __shared__ char smem[];
    __hip_bfloat16* Xlds = (__hip_bfloat16*)smem;                       // [6][68][32]
    __hip_bfloat16* Wlds = (__hip_bfloat16*)(smem + 6 * 68 * 32 * 2);   // [5][128][32]

    const int b    = blockIdx.x >> 5;
    const int tile = blockIdx.x & 31;
    const int h0   = tile * 2;
    const int t    = threadIdx.x;
    const int wave = t >> 5, lane = t & 31;
    const int wm = wave >> 1, wn = wave & 1;     // wm:0..3, wn:0..1
    const int lhalf = lane >> 4, l16 = lane & 15;

    v8f acc[2][4];
    for (int am = 0; am < 2; ++am)
        for (int an = 0; an < 4; ++an)
            for (int v = 0; v < 8; ++v) acc[am][an][v] = 0.f;

    for (int ci = 0; ci < 4; ++ci) {
        __syncthreads();  // previous compute done before overwriting Xlds
        // ---- stage X tile with halo: 6 rows x 68 cols x 32 ch, zero-padded.
        //      One 16B segment per lane: 6*68 chunks x 4 segments.
        for (int seg = t; seg < 6 * 68 * 4; seg += 256) {
            const int chunk = seg >> 2, q = seg & 3;
            const int lrow = chunk / 68, lcol = chunk % 68;
            const int gh = h0 - 2 + lrow, gw = lcol - 2;
            __hip_bfloat16* dst = &Xlds[chunk * 32 + q * 8];
            if (gh >= 0 && gh < HH && gw >= 0 && gw < WW) {
                const __hip_bfloat16* src =
                    &xh[((size_t)(b * HW + gh * WW + gw)) * CI + ci * 32 + q * 8];
                async_copy16(src, dst);
            } else {
                U4 z; z.x = z.y = z.z = z.w = 0u;
                *(U4*)dst = z;
            }
        }
        async_join();
        for (int r = 0; r < 5; ++r) {
            __syncthreads();  // Xlds/Wlds published / previous compute done
            // ---- stage W: 5 taps x 128 o x 32 ch for this (r, ci): 640 chunks x 4 segs
            for (int seg = t; seg < 5 * 128 * 4; seg += 256) {
                const int chunk = seg >> 2, q = seg & 3;
                const int s = chunk >> 7, o = chunk & 127;
                const __hip_bfloat16* sp =
                    &weff[(((size_t)((r * 5 + s) * CO + o) * BN + b)) * CI + ci * 32 + q * 8];
                async_copy16(sp, &Wlds[chunk * 32 + q * 8]);
                if (r < 4) // prefetch next tap-row's weights into cache
                    __builtin_prefetch(sp + (size_t)5 * CO * BN * CI, 0, 1);
            }
            async_join();
            __syncthreads();
            for (int s = 0; s < 5; ++s) {
                // ---- B fragments: 4 N-tiles; lane: n = l16, K = lhalf*16 + [0..15]
                v16bf bfrag[4];
                for (int an = 0; an < 4; ++an) {
                    const int o = (wn * 4 + an) * 16 + l16;
                    const U4* u = (const U4*)&Wlds[(s * 128 + o) * 32];
                    union { U4 u[2]; v16bf v; } tmp;
                    tmp.u[0] = u[lhalf * 2];
                    tmp.u[1] = u[lhalf * 2 + 1];
                    bfrag[an] = tmp.v;
                }
                // ---- A fragments: 2 M-tiles; lane: m = l16, K = lhalf*16 + [0..15]
                v16bf afrag[2];
                for (int am = 0; am < 2; ++am) {
                    const int p = (wm * 2 + am) * 16 + l16;
                    const int lrow = (p >> 6) + r;
                    const int lcol = (p & 63) + s;
                    const U4* u = (const U4*)&Xlds[(lrow * 68 + lcol) * 32];
                    union { U4 u[2]; v16bf v; } tmp;
                    tmp.u[0] = u[lhalf * 2];
                    tmp.u[1] = u[lhalf * 2 + 1];
                    afrag[am] = tmp.v;
                }
                for (int am = 0; am < 2; ++am)
                    for (int an = 0; an < 4; ++an)
                        acc[am][an] = __builtin_amdgcn_wmma_f32_16x16x32_bf16(
                            false, afrag[am], false, bfrag[an],
                            (short)0, acc[am][an], false, false);
            }
        }
    }

    // ---- epilogue: + effective bias, store NCHW f32
    for (int am = 0; am < 2; ++am) {
        for (int an = 0; an < 4; ++an) {
            const int o = (wn * 4 + an) * 16 + l16;
            const float be = bias_eff[b * CO + o];
            for (int v = 0; v < 8; ++v) {
                const int p = (wm * 2 + am) * 16 + v + lhalf * 8;
                const int hh = h0 + (p >> 6), ww = p & 63;
                out[((size_t)(b * CO + o)) * HW + hh * WW + ww] = acc[am][an][v] + be;
            }
        }
    }
}

// ---------------------------------------------------------------------------
extern "C" void kernel_launch(void* const* d_in, const int* in_sizes, int n_in,
                              void* d_out, int out_size, void* d_ws, size_t ws_size,
                              hipStream_t stream) {
    const float* x      = (const float*)d_in[0];
    const float* weight = (const float*)d_in[1];
    const float* bias   = (const float*)d_in[2];
    const float* a1w    = (const float*)d_in[3];
    const float* a1b    = (const float*)d_in[4];
    const float* a2w    = (const float*)d_in[5];
    const float* a2b    = (const float*)d_in[6];
    float* out = (float*)d_out;

    char* ws = (char*)d_ws;
    __hip_bfloat16* xh   = (__hip_bfloat16*)ws;                   // 33,554,432 B
    __hip_bfloat16* weff = (__hip_bfloat16*)(ws + 33554432);      // 26,214,400 B
    float* wt            = (float*)(ws + 59768832);               //  6,553,600 B
    float* pooled        = (float*)(ws + 66322432);               //     16,384 B
    float* attn          = (float*)(ws + 66338816);               //        512 B
    float* bias_eff      = (float*)(ws + 66339328);               //     16,384 B

    pool_kernel<<<BN * CI, 256, 0, stream>>>(x, pooled);
    attn_kernel<<<BN, 128, 0, stream>>>(pooled, a1w, a1b, a2w, a2b, bias, attn, bias_eff);
    convert_x<<<BN * 64 * 4, 256, 0, stream>>>(x, xh);
    transpose_w<<<(KB * TAPS * CO * CI + 255) / 256, 256, 0, stream>>>(weight, wt);
    combine_w<<<TAPS * CO, 128, 0, stream>>>(wt, attn, weff);

    const size_t lds_bytes = (size_t)(6 * 68 * 32 + 5 * 128 * 32) * 2;  // 67,072 B
    conv_wmma<<<BN * 32, 256, lds_bytes, stream>>>(xh, weff, bias_eff, out);
}
